// AdaptivePointMatcher_29437705847327
// MI455X (gfx1250) — matrix operand order repaired and checked by reading, not compiled, not run on hardware
//
#include <hip/hip_runtime.h>
#include <hip/hip_bf16.h>

typedef __attribute__((ext_vector_type(16))) _Float16 v16h;
typedef __attribute__((ext_vector_type(8)))  float    v8f;

#define NQ 1024   // N pred
#define MG 512    // M gt
#define DP 128    // D
#define K1 256    // 2D
#define H2 128    // second hidden

// ---------------------------------------------------------------------------
// Kernel 0: convert mn_w2 (256x128 f32, row-major [k][col]) into f16 WMMA
// B-fragments. Fragment (ks, ct) covers K = ks*32..+31, cols ct*16..+15.
// Per ISA 16-bit B 32x16 layout: lane l holds col = ct*16 + (l&15);
// halfs j=0..15 hold K_local = (j&7) + (j>=8 ? 16 : 0) + (l>=16 ? 8 : 0).
// Stored flat: w2f[(((ks*8+ct)*32)+lane)*16 + j]
// ---------------------------------------------------------------------------
__global__ void k_w2cvt(const float* __restrict__ mn_w2, _Float16* __restrict__ w2f) {
    int idx  = blockIdx.x * 256 + threadIdx.x;       // 0..32767
    int j    = idx & 15;
    int lane = (idx >> 4) & 31;
    int ct   = (idx >> 9) & 7;
    int ks   = idx >> 12;
    int k    = ks * 32 + (j & 7) + ((j & 8) ? 16 : 0) + ((lane & 16) ? 8 : 0);
    int col  = ct * 16 + (lane & 15);
    w2f[idx] = (_Float16)mn_w2[k * H2 + col];
}

// ---------------------------------------------------------------------------
// Kernel 1: point encoder. One block (128 thr) per row (1024 pred + 512 gt).
// feats = mean_p( relu(x @ w1 + b1) @ w2 ) + b2
// ---------------------------------------------------------------------------
__global__ void k_encode(const float* __restrict__ pred_pts, const float* __restrict__ gt_pts,
                         const float* __restrict__ w1, const float* __restrict__ b1,
                         const float* __restrict__ w2, const float* __restrict__ b2,
                         float* __restrict__ predf, float* __restrict__ gtf) {
    int row = blockIdx.x, t = threadIdx.x;           // t in [0,128)
    const float* base;
    float* out;
    if (row < NQ) { base = pred_pts + row * 40;          out = predf + row * DP; }
    else          { base = gt_pts + (row - NQ) * 40;     out = gtf  + (row - NQ) * DP; }

    __shared__ float h[64];
    float acc = 0.f;
    for (int p = 0; p < 20; ++p) {
        float px = base[p * 2], py = base[p * 2 + 1];
        if (t < 64) h[t] = fmaxf(px * w1[t] + py * w1[64 + t] + b1[t], 0.f);
        __syncthreads();
        float s = 0.f;
        #pragma unroll 8
        for (int j = 0; j < 64; ++j) s += h[j] * w2[j * DP + t];
        acc += s;
        __syncthreads();
    }
    out[t] = acc * (1.0f / 20.0f) + b2[t];
}

// ---------------------------------------------------------------------------
// Kernel 2: A256[n] = pred_f[n] @ W1_top ; B256[m] = gt_f[m] @ W1_bot + b1.
// One block (256 thr) per row (1024 + 512).
// ---------------------------------------------------------------------------
__global__ void k_ab(const float* __restrict__ predf, const float* __restrict__ gtf,
                     const float* __restrict__ mn_w1, const float* __restrict__ mn_b1,
                     float* __restrict__ A256, float* __restrict__ B256) {
    int row = blockIdx.x, t = threadIdx.x;           // t in [0,256)
    __shared__ float f[DP];
    const float* src; const float* w; float* out; float bias = 0.f;
    if (row < NQ) { src = predf + row * DP; w = mn_w1;              out = A256 + row * K1; }
    else { int m = row - NQ; src = gtf + m * DP; w = mn_w1 + DP * K1; out = B256 + m * K1; bias = mn_b1[t]; }
    if (t < DP) f[t] = src[t];
    __syncthreads();
    float s = bias;
    #pragma unroll 8
    for (int k = 0; k < DP; ++k) s += f[k] * w[k * K1 + t];
    out[t] = s;
}

// ---------------------------------------------------------------------------
// Kernel 3: main pair MLP + softmax + matched. One block (8 waves) per n.
// W2 fragments staged in LDS (64 KB), re-used (aliased) for softmax buffers.
// Per wave: 4 m-tiles of 16 rows; build 8 A-fragments (relu(a+b) -> f16) per
// tile, then ct-outer / ks-inner WMMA chain with one accumulator live.
// ---------------------------------------------------------------------------
__global__ __launch_bounds__(256) void k_match(
    const float* __restrict__ A256, const float* __restrict__ B256,
    const _Float16* __restrict__ w2f,
    const float* __restrict__ mn_b2, const float* __restrict__ mn_w3,
    const float* __restrict__ mn_b3, const float* __restrict__ gt_pts,
    float* __restrict__ out_matched, float* __restrict__ out_conf,
    float* __restrict__ out_probs) {

    int n   = blockIdx.x;
    int tid = threadIdx.x;
    int l   = tid & 31;          // lane in wave
    int wv  = tid >> 5;          // wave 0..7
    int rl  = l & 15;            // N-col / M-row sub-index
    int koff = (l & 16) ? 8 : 0; // A-fragment K offset for upper half-wave

    // 64 KB: W2 f16 fragments during GEMM; aliased as scores/reduction after.
    __shared__ _Float16 w2s[64 * 32 * 16];

    // cooperative copy: 32768 halfs = 4096 x 16B; 16 per thread
    {
        const uint4* src = (const uint4*)w2f;
        uint4*       dst = (uint4*)w2s;
        #pragma unroll
        for (int i = 0; i < 16; ++i)
            dst[tid + i * 256] = src[tid + i * 256];
    }
    __syncthreads();

    const float* arow = A256 + n * K1;
    float b2r[8], w3r[8];
    #pragma unroll
    for (int ct = 0; ct < 8; ++ct) {
        b2r[ct] = mn_b2[ct * 16 + rl];
        w3r[ct] = mn_w3[ct * 16 + rl];
    }
    float b3 = mn_b3[0];

    float fin[4][8];

    for (int t4 = 0; t4 < 4; ++t4) {
        __asm__ volatile("" ::: "memory");   // keep fragment loads inside the loop
        int m0 = (wv + t4 * 8) * 16;
        const float* brow = B256 + (m0 + rl) * K1;

        // Build all 8 A-fragments for this m-tile: afr[ks] = relu(a+b) as f16.
        v16h afr[8];
        #pragma unroll
        for (int ks = 0; ks < 8; ++ks) {
            int kb = ks * 32 + koff;
            float av[16], bv[16];
            *(float4*)(av + 0)  = *(const float4*)(arow + kb);
            *(float4*)(av + 4)  = *(const float4*)(arow + kb + 4);
            *(float4*)(av + 8)  = *(const float4*)(arow + kb + 16);
            *(float4*)(av + 12) = *(const float4*)(arow + kb + 20);
            *(float4*)(bv + 0)  = *(const float4*)(brow + kb);
            *(float4*)(bv + 4)  = *(const float4*)(brow + kb + 4);
            *(float4*)(bv + 8)  = *(const float4*)(brow + kb + 16);
            *(float4*)(bv + 12) = *(const float4*)(brow + kb + 20);
            #pragma unroll
            for (int j = 0; j < 16; ++j)
                afr[ks][j] = (_Float16)fmaxf(av[j] + bv[j], 0.f);
        }

        float partial[8];
        #pragma unroll
        for (int v = 0; v < 8; ++v) partial[v] = 0.f;

        #pragma unroll
        for (int ct = 0; ct < 8; ++ct) {
            v8f acc = (v8f){0.f,0.f,0.f,0.f,0.f,0.f,0.f,0.f};
            #pragma unroll
            for (int ks = 0; ks < 8; ++ks) {
                v16h bf = *(const v16h*)(w2s + (((ks * 8 + ct) * 32) + l) * 16);
                acc = __builtin_amdgcn_wmma_f32_16x16x32_f16(
                    false, afr[ks], false, bf, (short)0, acc, false, false);
            }
            // fold into score partials immediately (acc dies here)
            #pragma unroll
            for (int v = 0; v < 8; ++v)
                partial[v] += fmaxf(acc[v] + b2r[ct], 0.f) * w3r[ct];
        }

        // reduce across the 16 lanes of each half-wave (cols)
        #pragma unroll
        for (int v = 0; v < 8; ++v) {
            partial[v] += __shfl_xor(partial[v], 1, 32);
            partial[v] += __shfl_xor(partial[v], 2, 32);
            partial[v] += __shfl_xor(partial[v], 4, 32);
            partial[v] += __shfl_xor(partial[v], 8, 32);
            fin[t4][v] = partial[v] + b3;
        }
    }

    __syncthreads();                 // all waves done reading w2s
    float* sc  = (float*)w2s;        // [512] scores
    float* red = sc + MG;            // [256] reduction buffer

    if (rl == 0) {
        #pragma unroll
        for (int t4 = 0; t4 < 4; ++t4) {
            int rbase = (wv + t4 * 8) * 16 + ((l & 16) ? 8 : 0);
            #pragma unroll
            for (int v = 0; v < 8; ++v) sc[rbase + v] = fin[t4][v];
        }
    }
    __syncthreads();

    // softmax over 512 scores
    float s0 = sc[tid], s1 = sc[tid + 256];
    red[tid] = fmaxf(s0, s1);
    __syncthreads();
    for (int s = 128; s > 0; s >>= 1) {
        if (tid < s) red[tid] = fmaxf(red[tid], red[tid + s]);
        __syncthreads();
    }
    float smax = red[0];
    __syncthreads();
    float e0 = __expf(s0 - smax), e1 = __expf(s1 - smax);
    red[tid] = e0 + e1;
    __syncthreads();
    for (int s = 128; s > 0; s >>= 1) {
        if (tid < s) red[tid] += red[tid + s];
        __syncthreads();
    }
    float inv = 1.0f / red[0];
    __syncthreads();
    float p0 = e0 * inv, p1 = e1 * inv;
    sc[tid] = p0; sc[tid + 256] = p1;
    out_probs[n * MG + tid]       = p0;
    out_probs[n * MG + tid + 256] = p1;
    if (tid == 0) out_conf[n] = inv;   // max prob = exp(0)/sum = 1/sum
    __syncthreads();

    // matched[n][j] = sum_m probs[m] * gt_pts_flat[m][j], j in [0,40)
    if (tid < 40) {
        float s = 0.f;
        for (int m = 0; m < MG; ++m) s += sc[m] * gt_pts[m * 40 + tid];
        out_matched[n * 40 + tid] = s;
    }
}

// ---------------------------------------------------------------------------
extern "C" void kernel_launch(void* const* d_in, const int* in_sizes, int n_in,
                              void* d_out, int out_size, void* d_ws, size_t ws_size,
                              hipStream_t stream) {
    const float* pred_pts = (const float*)d_in[0];   // (1024,20,2)
    const float* gt_pts   = (const float*)d_in[1];   // (512,20,2)
    const float* pe_w1    = (const float*)d_in[2];   // (2,64)
    const float* pe_b1    = (const float*)d_in[3];   // (64)
    const float* pe_w2    = (const float*)d_in[4];   // (64,128)
    const float* pe_b2    = (const float*)d_in[5];   // (128)
    const float* mn_w1    = (const float*)d_in[6];   // (256,256)
    const float* mn_b1    = (const float*)d_in[7];   // (256)
    const float* mn_w2    = (const float*)d_in[8];   // (256,128)
    const float* mn_b2    = (const float*)d_in[9];   // (128)
    const float* mn_w3    = (const float*)d_in[10];  // (128,1)
    const float* mn_b3    = (const float*)d_in[11];  // (1)

    char* ws = (char*)d_ws;
    _Float16* w2f  = (_Float16*)ws;                          // 64 KB
    float*    predf = (float*)(ws + 65536);                  // 1024*128*4 = 512 KB
    float*    gtf   = predf + NQ * DP;                       // 512*128*4  = 256 KB
    float*    A256  = gtf + MG * DP;                         // 1024*256*4 = 1 MB
    float*    B256  = A256 + NQ * K1;                        // 512*256*4  = 512 KB

    float* out         = (float*)d_out;
    float* out_matched = out;                 // 1024*40
    float* out_conf    = out + NQ * 40;       // 1024
    float* out_probs   = out_conf + NQ;       // 1024*512

    k_w2cvt <<<128, 256, 0, stream>>>(mn_w2, w2f);
    k_encode<<<NQ + MG, 128, 0, stream>>>(pred_pts, gt_pts, pe_w1, pe_b1, pe_w2, pe_b2, predf, gtf);
    k_ab    <<<NQ + MG, 256, 0, stream>>>(predf, gtf, mn_w1, mn_b1, A256, B256);
    k_match <<<NQ, 256, 0, stream>>>(A256, B256, w2f, mn_b2, mn_w3, mn_b3, gt_pts,
                                     out_matched, out_conf, out_probs);
}
